// LongContextAttention_2491081032382
// MI455X (gfx1250) — compile-verified, matrix operand order checked by reading, not run on hardware
//
#include <hip/hip_runtime.h>

// ---------------------------------------------------------------------------
// Exact softmax attention (flash style), B=2 S=2048 H=16 D=128 fp32.
// Compute-bound (~69 GFLOP vs ~134 MB) -> v_wmma_f32_16x16x32_bf16.
// TRANSPOSED formulation: S^T = K Q^T (queries = WMMA columns = lanes),
// O^T = V^T P^T. Softmax stats are per-lane scalars (1 shuffle each);
// P^T B-operand packs straight from registers (no LDS transpose); output
// stores as b128. K staged row-major, V staged transposed (d-major) in LDS
// as bf16; 64-key steps; next tile's global loads issue in the barrier
// shadow (split s_barrier_signal/-wait). Scores are computed in the log2
// domain (1/sqrt(D)*log2e folded into Q) so softmax uses raw v_exp_f32
// via __builtin_amdgcn_exp2f.
// ---------------------------------------------------------------------------

typedef __attribute__((ext_vector_type(16))) __bf16   v16bf;
typedef __attribute__((ext_vector_type(8)))  float    v8f;
typedef __attribute__((ext_vector_type(4)))  float    f32x4;
typedef __attribute__((ext_vector_type(4)))  unsigned u32x4;
typedef __attribute__((ext_vector_type(2)))  unsigned u32x2;

#if __has_builtin(__builtin_amdgcn_s_barrier_signal) && __has_builtin(__builtin_amdgcn_s_barrier_wait)
#define SPLIT_BARRIER 1
#else
#define SPLIT_BARRIER 0
#endif

namespace {
constexpr int BB = 2, SS = 2048, HH = 16, DD = 128;
constexpr int QB   = 128;      // queries per block (16 per wave * 8 waves)
constexpr int KBLK = 64;       // keys per flash step (2x WMMA K)
constexpr int KSTR = DD + 8;   // lds_k row stride (bf16), 16B-aligned reads
constexpr int VSTR = KBLK + 8; // lds_vt row stride (bf16), 16B-aligned reads
// 1/sqrt(128) * log2(e): scores land in the exp2 domain.
constexpr float SCALE = 0.08838834764831845f * 1.44269504088896340736f;
}

__device__ __forceinline__ unsigned pack_bf2(float a, float b) {
  unsigned ua = __builtin_bit_cast(unsigned, a);
  unsigned ub = __builtin_bit_cast(unsigned, b);
  ua += 0x7FFFu + ((ua >> 16) & 1u);   // round-to-nearest-even to bf16
  ub += 0x7FFFu + ((ub >> 16) & 1u);
  return (ua >> 16) | (ub & 0xFFFF0000u);
}

// XOR-shuffle across the wave via ds_swizzle_b32 (group-of-32 mode).
template <int M>
__device__ __forceinline__ float swz(float v) {
  return __builtin_bit_cast(float,
      __builtin_amdgcn_ds_swizzle(__builtin_bit_cast(int, v),
                                  (M << 10) | 0x1F));
}

__device__ __forceinline__ v8f vmax8(v8f a, v8f b) {
  v8f r;
#pragma unroll
  for (int i = 0; i < 8; ++i) r[i] = fmaxf(a[i], b[i]);
  return r;
}

__global__ __launch_bounds__(256)
void fa_fwd_wmma(const float* __restrict__ Q, const float* __restrict__ K,
                 const float* __restrict__ V, float* __restrict__ O) {
  __shared__ __bf16 lds_k[KBLK * KSTR];  // [key][d]   (S^T A-operand)
  __shared__ __bf16 lds_vt[DD * VSTR];   // [d][key]   (O^T A-operand)

  const int tid   = threadIdx.x;
  const int wave  = tid >> 5;
  const int lane  = tid & 31;
  const int hsel  = lane >> 4;     // 16-lane half
  const int n     = lane & 15;     // query column owned by this lane
  const int kbase = hsel * 8;      // K-dim offset of this half in fragments

  const int b  = blockIdx.z, h = blockIdx.y;
  const int q0 = blockIdx.x * QB + wave * 16;
  const size_t rs = (size_t)HH * DD;   // seq-row stride in floats

  // ---- Q^T B-fragments (lane = query col n), pre-scaled ----
  const float* qrow = Q + ((size_t)(b * SS + q0 + n) * HH + h) * DD;
  v16bf bq[4];
#pragma unroll
  for (int t = 0; t < 4; ++t) {
    const float* p = qrow + t * 32 + kbase;
    f32x4 x0 = *(const f32x4*)(p);
    f32x4 x1 = *(const f32x4*)(p + 4);
    f32x4 x2 = *(const f32x4*)(p + 16);
    f32x4 x3 = *(const f32x4*)(p + 20);
    union { v16bf v; unsigned u[8]; } bf;
    bf.u[0] = pack_bf2(x0.x * SCALE, x0.y * SCALE);
    bf.u[1] = pack_bf2(x0.z * SCALE, x0.w * SCALE);
    bf.u[2] = pack_bf2(x1.x * SCALE, x1.y * SCALE);
    bf.u[3] = pack_bf2(x1.z * SCALE, x1.w * SCALE);
    bf.u[4] = pack_bf2(x2.x * SCALE, x2.y * SCALE);
    bf.u[5] = pack_bf2(x2.z * SCALE, x2.w * SCALE);
    bf.u[6] = pack_bf2(x3.x * SCALE, x3.y * SCALE);
    bf.u[7] = pack_bf2(x3.z * SCALE, x3.w * SCALE);
    bq[t] = bf.v;
  }

  v8f zero = {};
  v8f acc[8];                      // O^T: 8 tiles of (16 d x 16 queries)
#pragma unroll
  for (int dt = 0; dt < 8; ++dt) acc[dt] = zero;
  float mrun = -3.0e38f, lrun = 0.0f;   // per-lane (= per-query) stats

  const float* kg = K + ((size_t)b * SS * HH + h) * DD;
  const float* vg = V + ((size_t)b * SS * HH + h) * DD;
  const int d0 = lane * 4;

  // ---- software-pipelined staging registers (tile kb=0) ----
  f32x4 kf[8], vf[8];
#pragma unroll
  for (int it = 0; it < 8; ++it)
    kf[it] = *(const f32x4*)(kg + (size_t)(wave + 8 * it) * rs + d0);
#pragma unroll
  for (int i = 0; i < 4; ++i) {
    vf[i]     = *(const f32x4*)(vg + (size_t)(4 * wave + i) * rs + d0);
    vf[4 + i] = *(const f32x4*)(vg + (size_t)(32 + 4 * wave + i) * rs + d0);
  }

  for (int kb = 0; kb < SS; kb += KBLK) {
    // ---- pack + store the prefetched tile into LDS ----
#pragma unroll
    for (int it = 0; it < 8; ++it) {
      u32x2 pk;
      pk.x = pack_bf2(kf[it].x, kf[it].y);
      pk.y = pack_bf2(kf[it].z, kf[it].w);
      *(u32x2*)(&lds_k[(wave + 8 * it) * KSTR + d0]) = pk;
    }
#pragma unroll
    for (int dd = 0; dd < 4; ++dd) {     // two 4x4 register transposes of V
      u32x2 p0, p1;
      p0.x = pack_bf2(vf[0][dd], vf[1][dd]);
      p0.y = pack_bf2(vf[2][dd], vf[3][dd]);
      p1.x = pack_bf2(vf[4][dd], vf[5][dd]);
      p1.y = pack_bf2(vf[6][dd], vf[7][dd]);
      *(u32x2*)(&lds_vt[(d0 + dd) * VSTR + 4 * wave])      = p0;
      *(u32x2*)(&lds_vt[(d0 + dd) * VSTR + 32 + 4 * wave]) = p1;
    }

    // ---- barrier; prefetch next tile in its shadow ----
#if SPLIT_BARRIER
    asm volatile("s_wait_dscnt 0" ::: "memory");
    __builtin_amdgcn_s_barrier_signal(-1);
#else
    __syncthreads();
#endif
    int kbn = kb + KBLK;
    if (kbn >= SS) kbn = SS - KBLK;      // clamped; final prefetch discarded
#pragma unroll
    for (int it = 0; it < 8; ++it)
      kf[it] = *(const f32x4*)(kg + (size_t)(kbn + wave + 8 * it) * rs + d0);
#pragma unroll
    for (int i = 0; i < 4; ++i) {
      vf[i]     = *(const f32x4*)(vg + (size_t)(kbn + 4 * wave + i) * rs + d0);
      vf[4 + i] = *(const f32x4*)(vg + (size_t)(kbn + 32 + 4 * wave + i) * rs + d0);
    }
#if SPLIT_BARRIER
    __builtin_amdgcn_s_barrier_wait((short)-1);
    asm volatile("" ::: "memory");
#endif

    // ---- S^T = K Q^T : 4 tiles of (16 keys x 16 queries) ----
    v8f st[4] = {zero, zero, zero, zero};
#pragma unroll
    for (int hk = 0; hk < 4; ++hk) {
#pragma unroll
      for (int t = 0; t < 4; ++t) {
        const __bf16* kp = &lds_k[(hk * 16 + n) * KSTR + t * 32 + kbase];
        union { v16bf v; u32x4 u[2]; } ka;
        ka.u[0] = *(const u32x4*)(kp);
        ka.u[1] = *(const u32x4*)(kp + 16);
        st[hk] = __builtin_amdgcn_wmma_f32_16x16x32_bf16(
            false, ka.v, false, bq[t], (short)0, st[hk], false, false);
      }
    }

    // ---- online softmax (log2 domain): keys are rows -> scalar stats ----
    v8f mv = vmax8(vmax8(st[0], st[1]), vmax8(st[2], st[3]));
    float m0 = fmaxf(fmaxf(mv[0], mv[1]), fmaxf(mv[2], mv[3]));
    float m1 = fmaxf(fmaxf(mv[4], mv[5]), fmaxf(mv[6], mv[7]));
    float mt = fmaxf(m0, m1);
    mt = fmaxf(mt, swz<16>(mt));         // combine the two key-halves
    float mn = fmaxf(mrun, mt);
    float alpha = __builtin_amdgcn_exp2f(mrun - mn);
    mrun = mn;
#pragma unroll
    for (int hk = 0; hk < 4; ++hk)
#pragma unroll
      for (int j = 0; j < 8; ++j)
        st[hk][j] = __builtin_amdgcn_exp2f(st[hk][j] - mn);
    v8f sv = (st[0] + st[1]) + (st[2] + st[3]);
    float s0 = (sv[0] + sv[1]) + (sv[2] + sv[3]);
    float s1 = (sv[4] + sv[5]) + (sv[6] + sv[7]);
    float ssum = s0 + s1;
    ssum += swz<16>(ssum);
    lrun = lrun * alpha + ssum;
#pragma unroll
    for (int dt = 0; dt < 8; ++dt) acc[dt] = acc[dt] * alpha;

    // ---- P^T B-fragments: packed straight from registers ----
    union { v16bf v; unsigned u[8]; } bp01, bp23;
#pragma unroll
    for (int i = 0; i < 4; ++i) {
      bp01.u[i]     = pack_bf2(st[0][2 * i], st[0][2 * i + 1]);  // keys kbase+e
      bp01.u[4 + i] = pack_bf2(st[1][2 * i], st[1][2 * i + 1]);  // keys 16+kbase+e
      bp23.u[i]     = pack_bf2(st[2][2 * i], st[2][2 * i + 1]);  // keys 32+...
      bp23.u[4 + i] = pack_bf2(st[3][2 * i], st[3][2 * i + 1]);  // keys 48+...
    }

    // ---- O^T += V^T P^T : V^T rows contiguous in lds_vt ----
#pragma unroll
    for (int dt = 0; dt < 8; ++dt) {
      const __bf16* vp = &lds_vt[(dt * 16 + n) * VSTR + kbase];
      union { v16bf v; u32x4 u[2]; } va;
      va.u[0] = *(const u32x4*)(vp);         // keys kbase..,16+kbase..
      va.u[1] = *(const u32x4*)(vp + 16);
      acc[dt] = __builtin_amdgcn_wmma_f32_16x16x32_bf16(
          false, va.v, false, bp01.v, (short)0, acc[dt], false, false);
      union { v16bf v; u32x4 u[2]; } vb;
      vb.u[0] = *(const u32x4*)(vp + 32);    // keys 32+kbase..,48+kbase..
      vb.u[1] = *(const u32x4*)(vp + 48);
      acc[dt] = __builtin_amdgcn_wmma_f32_16x16x32_bf16(
          false, vb.v, false, bp23.v, (short)0, acc[dt], false, false);
    }
    __syncthreads();   // all waves done reading before next tile overwrites
  }

  // ---- normalize and store: lane = query row, 8 consecutive d per tile ----
  float inv = 1.0f / lrun;
  float* op = O + ((size_t)(b * SS + q0 + n) * HH + h) * DD + kbase;
#pragma unroll
  for (int dt = 0; dt < 8; ++dt) {
    v8f r = acc[dt] * inv;
    f32x4 lo = {r[0], r[1], r[2], r[3]};
    f32x4 hi = {r[4], r[5], r[6], r[7]};
    *(f32x4*)(op + dt * 16)     = lo;
    *(f32x4*)(op + dt * 16 + 4) = hi;
  }
}

extern "C" void kernel_launch(void* const* d_in, const int* in_sizes, int n_in,
                              void* d_out, int out_size, void* d_ws, size_t ws_size,
                              hipStream_t stream) {
  (void)in_sizes; (void)n_in; (void)out_size; (void)d_ws; (void)ws_size;
  const float* q = (const float*)d_in[0];
  const float* k = (const float*)d_in[1];
  const float* v = (const float*)d_in[2];
  float* o = (float*)d_out;
  dim3 grid(SS / QB, HH, BB);   // 16 x 16 x 2 = 512 workgroups
  dim3 block(256);              // 8 wave32
  fa_fwd_wmma<<<grid, block, 0, stream>>>(q, k, v, o);
}
